// PredictComsize_36103495090827
// MI455X (gfx1250) — matrix operand order repaired and checked by reading, not compiled, not run on hardware
//
#include <hip/hip_runtime.h>
#include <hip/hip_bf16.h>

// ---------------------------------------------------------------------------
// Graph-transformer forward for MI455X (gfx1250, wave32, WMMA).
//  - Dense projections: v_wmma_f32_16x16x32_f16, B staged in LDS via the
//    CDNA5 async-to-LDS path (ASYNCcnt), 64x16 strip per wave (16 WMMA/wave).
//  - Edge softmax/aggregation: L2-resident gathers + float atomics.
// ---------------------------------------------------------------------------

#define D_MODEL 128
#define HEADS   4
#define D_HEAD  32

typedef __attribute__((ext_vector_type(16))) _Float16 v16h;
typedef __attribute__((ext_vector_type(8)))  float    v8f;

union Frag16 { uint4 u[2]; v16h v; };
union Half8  { uint4 u; _Float16 h[8]; };

// ---------------------------------------------------------------------------
// Weight transpose + f32 -> f16 convert:  Wt[n][k] = W[k][n]
// ---------------------------------------------------------------------------
__global__ void xpose_cvt_kernel(const float* __restrict__ W,
                                 _Float16* __restrict__ Wt) {
    int k = blockIdx.x;    // 0..127
    int n = threadIdx.x;   // 0..127
    Wt[n * D_MODEL + k] = (_Float16)W[k * D_MODEL + n];
}

// ---------------------------------------------------------------------------
// h = X + Seed*W_seed + D*W_deg + C*W_clu   (rank-1 embeddings), store f16
// ---------------------------------------------------------------------------
__global__ void embed_kernel(const float* __restrict__ X,
                             const float* __restrict__ Seed,
                             const float* __restrict__ Dg,
                             const float* __restrict__ Cl,
                             const float* __restrict__ Ws,
                             const float* __restrict__ Wd,
                             const float* __restrict__ Wc,
                             _Float16* __restrict__ h) {
    int node = blockIdx.x;
    int c = threadIdx.x;
    float val = X[node] + Seed[node] * Ws[c] + Dg[node] * Wd[c] + Cl[node] * Wc[c];
    h[node * D_MODEL + c] = (_Float16)val;
}

// ---------------------------------------------------------------------------
// WMMA GEMM: C[M x 128] = A[M x 128] * B[128 x 128]
//   A: f16 row-major. Bt: f16, Bt[n][k] = B[k][n].
//   Block = 256 threads = 8 waves; block tile = 64 rows x 128 cols.
//   Whole Bt (32 KB) is async-copied to LDS once per block
//   (global_load_async_to_lds_b128 -> s_wait_asynccnt 0 -> barrier).
//   Wave w owns cols [16w,16w+16) and 4 row-tiles (4 accumulators), so each
//   LDS B-fragment is reused by 4 WMMAs. K=128 -> 16 WMMAs per wave.
//   MODE 0: f32 out   MODE 1: f16 out   MODE 2: f16 out with bias + ReLU
// ---------------------------------------------------------------------------
template <int MODE>
__global__ __launch_bounds__(256)
void gemm_f16_kernel(const _Float16* __restrict__ A,
                     const _Float16* __restrict__ Bt,
                     float* __restrict__ Cf,
                     _Float16* __restrict__ Ch,
                     const float* __restrict__ bias,
                     int M) {
    __shared__ _Float16 Bs[D_MODEL * D_MODEL];   // 32 KB, LDS offset 0

    const int tid  = threadIdx.x;
    const int lane = tid & 31;
    const int wave = tid >> 5;
    const int m0   = blockIdx.x * 64;
    const int n0   = wave * 16;
    const int half = lane >> 4;            // 0 or 1 (lane group)
    const int col  = n0 + (lane & 15);

    // ---- async-stage whole Bt into LDS (2048 x 16B chunks, 8 per thread) --
    {
        const char* gsrc = (const char*)Bt;
#pragma unroll
        for (int it = 0; it < 8; ++it) {
            unsigned byteoff = (unsigned)(it * 256 + tid) * 16u;
            unsigned ldsa = byteoff;   // Bs is the only LDS object -> offset 0
            unsigned long long ga = (unsigned long long)(gsrc + byteoff);
            asm volatile("global_load_async_to_lds_b128 %0, %1, off"
                         :: "v"(ldsa), "v"(ga) : "memory");
        }
        asm volatile("s_wait_asynccnt 0" ::: "memory");
    }
    __syncthreads();

    // ---- per-wave fragment pointers ----
    const _Float16* arow[4];
#pragma unroll
    for (int t = 0; t < 4; ++t) {
        int r = m0 + t * 16 + (lane & 15);
        if (r >= M) r = M - 1;             // tail-safe load (stores guarded)
        arow[t] = A + (size_t)r * D_MODEL + half * 16;
    }
    const _Float16* bls = Bs + (size_t)col * D_MODEL + half * 16;

    v8f acc[4] = {};
#pragma unroll
    for (int kt = 0; kt < 4; ++kt) {
        const int k0 = kt * 32;
        Frag16 bf;                          // from LDS (ds_load_b128)
        bf.u[0] = *(const uint4*)(bls + k0);        // K = k0+16h   .. +7
        bf.u[1] = *(const uint4*)(bls + k0 + 8);    // K = k0+16h+8 .. +7
#pragma unroll
        for (int t = 0; t < 4; ++t) {
            Frag16 af;                      // from global (L2-resident)
            af.u[0] = *(const uint4*)(arow[t] + k0);       // K = k0+16h
            af.u[1] = *(const uint4*)(arow[t] + k0 + 8);   // K = k0+16h+8
            acc[t] = __builtin_amdgcn_wmma_f32_16x16x32_f16(
                /*neg_a=*/false, af.v, /*neg_b=*/false, bf.v,
                /*c_mod=*/(short)0, acc[t], /*reuse_a=*/false, /*reuse_b=*/false);
        }
    }

    // ---- epilogue: single block-level tail test, straight-line fast path --
    if (m0 + 64 <= M) {
#pragma unroll
        for (int t = 0; t < 4; ++t) {
            const int rbase = m0 + t * 16 + half * 8;
#pragma unroll
            for (int r = 0; r < 8; ++r) {
                float v = acc[t][r];
                const size_t o = (size_t)(rbase + r) * D_MODEL + col;
                if (MODE == 0)      Cf[o] = v;
                else if (MODE == 1) Ch[o] = (_Float16)v;
                else {
                    v += bias[col];
                    Ch[o] = (_Float16)(v > 0.0f ? v : 0.0f);
                }
            }
        }
    } else {
#pragma unroll
        for (int t = 0; t < 4; ++t) {
            const int rbase = m0 + t * 16 + half * 8;
#pragma unroll
            for (int r = 0; r < 8; ++r) {
                const int m = rbase + r;
                if (m >= M) continue;
                float v = acc[t][r];
                const size_t o = (size_t)m * D_MODEL + col;
                if (MODE == 0)      Cf[o] = v;
                else if (MODE == 1) Ch[o] = (_Float16)v;
                else {
                    v += bias[col];
                    Ch[o] = (_Float16)(v > 0.0f ? v : 0.0f);
                }
            }
        }
    }
}

// ---------------------------------------------------------------------------
// Edge scores: s[e][h] = dot(q[dst], k[src]) / sqrt(Dh); segment-max via
// order-preserving uint encoding + atomicMax (m_enc pre-zeroed; 0 < enc(-inf))
// ---------------------------------------------------------------------------
__device__ __forceinline__ unsigned enc_f32(float f) {
    unsigned u = __float_as_uint(f);
    return (u & 0x80000000u) ? ~u : (u | 0x80000000u);
}
__device__ __forceinline__ float dec_f32(unsigned key) {
    return (key & 0x80000000u) ? __uint_as_float(key & 0x7FFFFFFFu)
                               : __uint_as_float(~key);
}

__global__ void edge_score_kernel(const float* __restrict__ q,
                                  const float* __restrict__ k,
                                  const int* __restrict__ src,
                                  const int* __restrict__ dst,
                                  float* __restrict__ scores,
                                  unsigned* __restrict__ m_enc,
                                  int E_) {
    int idx = blockIdx.x * blockDim.x + threadIdx.x;
    if (idx >= E_ * HEADS) return;
    int e  = idx >> 2;
    int hh = idx & (HEADS - 1);
    int s = src[e], d = dst[e];
    const float4* qp = (const float4*)(q + (size_t)d * D_MODEL + hh * D_HEAD);
    const float4* kp = (const float4*)(k + (size_t)s * D_MODEL + hh * D_HEAD);
    float acc = 0.0f;
#pragma unroll
    for (int i = 0; i < D_HEAD / 4; ++i) {
        float4 a = qp[i], b = kp[i];
        acc += a.x * b.x + a.y * b.y + a.z * b.z + a.w * b.w;
    }
    acc *= 0.17677669529663687f;   // 1/sqrt(32)
    scores[idx] = acc;
    atomicMax(m_enc + (size_t)d * HEADS + hh, enc_f32(acc));
}

// ---------------------------------------------------------------------------
// ex = exp(s - m[dst]); denom[dst] += ex   (in-place overwrite of scores)
// ---------------------------------------------------------------------------
__global__ void edge_exp_kernel(float* __restrict__ scores,
                                const int* __restrict__ dst,
                                const unsigned* __restrict__ m_enc,
                                float* __restrict__ denom,
                                int E_) {
    int idx = blockIdx.x * blockDim.x + threadIdx.x;
    if (idx >= E_ * HEADS) return;
    int e  = idx >> 2;
    int hh = idx & (HEADS - 1);
    int d = dst[e];
    float m = dec_f32(m_enc[(size_t)d * HEADS + hh]);
    if (!(m >= -3.4e38f && m <= 3.4e38f)) m = 0.0f;   // empty-segment guard
    float ex = __expf(scores[idx] - m);
    scores[idx] = ex;
    atomicAdd(denom + (size_t)d * HEADS + hh, ex);
}

// ---------------------------------------------------------------------------
// agg[dst][c] += (ex / (denom[dst]+1e-9)) * v[src][c]   (one thread per e,c)
// ---------------------------------------------------------------------------
__global__ void edge_agg_kernel(const float* __restrict__ ex,
                                const float* __restrict__ denom,
                                const float* __restrict__ v,
                                const int* __restrict__ src,
                                const int* __restrict__ dst,
                                float* __restrict__ agg,
                                long long total) {
    long long idx = (long long)blockIdx.x * blockDim.x + threadIdx.x;
    if (idx >= total) return;
    int e   = (int)(idx >> 7);
    int c   = (int)(idx & (D_MODEL - 1));
    int hh  = c >> 5;
    int s = src[e], d = dst[e];
    float a = ex[(size_t)e * HEADS + hh] /
              (denom[(size_t)d * HEADS + hh] + 1e-9f);
    atomicAdd(agg + (size_t)d * D_MODEL + c, a * v[(size_t)s * D_MODEL + c]);
}

// ---------------------------------------------------------------------------
// f32 -> f16 convert
// ---------------------------------------------------------------------------
__global__ void cvt_f32_f16_kernel(const float* __restrict__ in,
                                   _Float16* __restrict__ out, int n) {
    int i = blockIdx.x * blockDim.x + threadIdx.x;
    if (i < n) out[i] = (_Float16)in[i];
}

// ---------------------------------------------------------------------------
// Output head: out[n] = dot(z[n], W2) + b2  (16B vector reads of z)
// ---------------------------------------------------------------------------
__global__ void head_kernel(const _Float16* __restrict__ z,
                            const float* __restrict__ W2,
                            const float* __restrict__ b2,
                            float* __restrict__ out, int n) {
    int node = blockIdx.x * blockDim.x + threadIdx.x;
    if (node >= n) return;
    const _Float16* zp = z + (size_t)node * D_MODEL;
    float acc = 0.0f;
#pragma unroll
    for (int cc = 0; cc < D_MODEL / 8; ++cc) {
        Half8 hx;
        hx.u = *(const uint4*)(zp + cc * 8);
#pragma unroll
        for (int j = 0; j < 8; ++j) acc += (float)hx.h[j] * W2[cc * 8 + j];
    }
    out[node] = acc + b2[0];
}

// ---------------------------------------------------------------------------
// Orchestration
// ---------------------------------------------------------------------------
extern "C" void kernel_launch(void* const* d_in, const int* in_sizes, int n_in,
                              void* d_out, int out_size, void* d_ws, size_t ws_size,
                              hipStream_t stream) {
    const float* X    = (const float*)d_in[0];
    const float* Seed = (const float*)d_in[1];
    const float* Dg   = (const float*)d_in[2];
    const float* Cl   = (const float*)d_in[3];
    const int*   src  = (const int*)d_in[4];
    const int*   dst  = (const int*)d_in[5];
    const float* Wse  = (const float*)d_in[6];
    const float* Wde  = (const float*)d_in[7];
    const float* Wcl  = (const float*)d_in[8];
    const float* Wq   = (const float*)d_in[9];   // [3,128,128]
    const float* Wk   = (const float*)d_in[10];
    const float* Wv   = (const float*)d_in[11];
    const float* Wo   = (const float*)d_in[12];
    const float* W1   = (const float*)d_in[13];
    const float* b1   = (const float*)d_in[14];
    const float* W2   = (const float*)d_in[15];
    const float* b2   = (const float*)d_in[16];

    const int N = in_sizes[0];        // 50000 nodes
    const int E = in_sizes[4];        // 1600000 edges
    const size_t ND = (size_t)N * D_MODEL;
    const size_t MAT = (size_t)D_MODEL * D_MODEL;

    // ---- workspace carve-out (256B aligned) ----
    char* ws = (char*)d_ws;
    size_t off = 0;
    auto carve = [&](size_t bytes) -> void* {
        void* p = ws + off;
        off = (off + bytes + 255) & ~(size_t)255;
        return p;
    };
    _Float16* h16    = (_Float16*)carve(ND * 2);
    float*    qbuf   = (float*)   carve(ND * 4);
    float*    kbuf   = (float*)   carve(ND * 4);
    float*    vbuf   = (float*)   carve(ND * 4);
    float*    agg    = (float*)   carve(ND * 4);
    _Float16* agg16  = (_Float16*)carve(ND * 2);
    float*    scores = (float*)   carve((size_t)E * HEADS * 4);
    unsigned* m_enc  = (unsigned*)carve((size_t)N * HEADS * 4);
    float*    denom  = (float*)   carve((size_t)N * HEADS * 4);
    _Float16* z16    = (_Float16*)carve(ND * 2);
    _Float16* Wt[13];
    for (int i = 0; i < 13; ++i) Wt[i] = (_Float16*)carve(MAT * 2);
    (void)ws_size; (void)n_in; (void)out_size;

    // ---- weights: transpose + convert to f16 (Bt[n][k] = B[k][n]) ----
    for (int l = 0; l < 3; ++l) {
        xpose_cvt_kernel<<<D_MODEL, D_MODEL, 0, stream>>>(Wq + l * MAT, Wt[l * 4 + 0]);
        xpose_cvt_kernel<<<D_MODEL, D_MODEL, 0, stream>>>(Wk + l * MAT, Wt[l * 4 + 1]);
        xpose_cvt_kernel<<<D_MODEL, D_MODEL, 0, stream>>>(Wv + l * MAT, Wt[l * 4 + 2]);
        xpose_cvt_kernel<<<D_MODEL, D_MODEL, 0, stream>>>(Wo + l * MAT, Wt[l * 4 + 3]);
    }
    xpose_cvt_kernel<<<D_MODEL, D_MODEL, 0, stream>>>(W1, Wt[12]);

    // ---- input embedding ----
    embed_kernel<<<N, D_MODEL, 0, stream>>>(X, Seed, Dg, Cl, Wse, Wde, Wcl, h16);

    const int gemmGrid = (N + 63) / 64;
    const int eThreads = E * HEADS;
    const int eBlocks  = (eThreads + 255) / 256;
    const long long aggTotal  = (long long)E * D_MODEL;
    const int aggBlocks = (int)((aggTotal + 255) / 256);

    for (int l = 0; l < 3; ++l) {
        // q, k, v projections (WMMA)
        gemm_f16_kernel<0><<<gemmGrid, 256, 0, stream>>>(h16, Wt[l * 4 + 0], qbuf, nullptr, nullptr, N);
        gemm_f16_kernel<0><<<gemmGrid, 256, 0, stream>>>(h16, Wt[l * 4 + 1], kbuf, nullptr, nullptr, N);
        gemm_f16_kernel<0><<<gemmGrid, 256, 0, stream>>>(h16, Wt[l * 4 + 2], vbuf, nullptr, nullptr, N);

        // segment softmax state
        hipMemsetAsync(m_enc, 0, (size_t)N * HEADS * 4, stream);
        hipMemsetAsync(denom, 0, (size_t)N * HEADS * 4, stream);
        hipMemsetAsync(agg,   0, ND * 4, stream);

        edge_score_kernel<<<eBlocks, 256, 0, stream>>>(qbuf, kbuf, src, dst, scores, m_enc, E);
        edge_exp_kernel<<<eBlocks, 256, 0, stream>>>(scores, dst, m_enc, denom, E);
        edge_agg_kernel<<<aggBlocks, 256, 0, stream>>>(scores, denom, vbuf, src, dst, agg, aggTotal);

        // output projection: h <- agg @ Wo (WMMA, f16 out)
        cvt_f32_f16_kernel<<<(int)((ND + 255) / 256), 256, 0, stream>>>(agg, agg16, (int)ND);
        gemm_f16_kernel<1><<<gemmGrid, 256, 0, stream>>>(agg16, Wt[l * 4 + 3], nullptr, h16, nullptr, N);
    }

    // MLP head: z = relu(h @ W1 + b1) (WMMA + fused epilogue), out = z @ W2 + b2
    gemm_f16_kernel<2><<<gemmGrid, 256, 0, stream>>>(h16, Wt[12], nullptr, z16, b1, N);
    head_kernel<<<(N + 255) / 256, 256, 0, stream>>>(z16, W2, b2, (float*)d_out, N);
}
